// GroupQueryAttention_52905407152829
// MI455X (gfx1250) — matrix operand order repaired and checked by reading, not compiled, not run on hardware
//
#include <hip/hip_runtime.h>

// Sizes fixed by the reference problem.
#define B_   8
#define C_   512
#define N_   1024      // H*W
#define G_   8
#define D_   128
#define Cg_  64
#define E_   1024      // G_*D_ == 2C
#define EPS  1e-5f

typedef __attribute__((ext_vector_type(16))) _Float16 v16h;
typedef __attribute__((ext_vector_type(8)))  float    v8f;

__device__ __forceinline__ v8f wmma16(v16h a, v16h b, v8f c) {
  // D = A(16x32 f16) * B(32x16 f16) + C(16x16 f32)
  return __builtin_amdgcn_wmma_f32_16x16x32_f16(false, a, false, b, (short)0, c,
                                                false, false);
}

__device__ __forceinline__ float gelu_exact(float x) {
  return 0.5f * x * (1.0f + erff(x * 0.7071067811865476f));
}

// ---------------------------------------------------------------------------
// Kernel 1: grouped 1x1-conv projections q,k,v + BN + GELU, stored n-major f16
//   per (b,g):  Y[128, 1024] = W[128,64] @ X[64,1024];  out[(bg,n,d)] = f16
// grid = (B*G, N/128), block = 256 (8 waves); each wave: 128(d) x 16(n)
// ---------------------------------------------------------------------------
__global__ __launch_bounds__(256)
void proj_qkv_kernel(const float* __restrict__ x,
                     const float* __restrict__ wq, const float* __restrict__ bq, const float* __restrict__ bnq,
                     const float* __restrict__ wk, const float* __restrict__ bk, const float* __restrict__ bnk,
                     const float* __restrict__ wv, const float* __restrict__ bv, const float* __restrict__ bnv,
                     _Float16* __restrict__ qT, _Float16* __restrict__ kT, _Float16* __restrict__ vT)
{
  const int bg  = blockIdx.x;               // b*G + g
  const int g   = bg & (G_ - 1);
  const int n0  = blockIdx.y * 128;
  const int tid = threadIdx.x;
  const int w   = tid >> 5, lane = tid & 31;
  const int mlane = lane & 15;              // A-frag row within tile
  const int kbA   = (lane >> 4) << 3;       // unused (0|8); A-frag uses kbB
  const int kbB   = (lane >> 4) << 4;       // B-frag k offset (0|16)
  const int mbase = (lane >> 4) << 3;       // C-frag row base (0|8)
  const int ncol  = lane & 15;              // C/B-frag column
  const int nw    = n0 + w * 16;

  // x layout [B,C,N]: (b*C + g*64)*N == bg*64*N
  const float* xg = x + (size_t)bg * Cg_ * N_;

  const float* wmat[3] = {wq, wk, wv};
  const float* bias[3] = {bq, bk, bv};
  const float* bnm[3]  = {bnq, bnk, bnv};
  _Float16*    outp[3] = {qT, kT, vT};

  const v8f vz = {};

  #pragma unroll
  for (int p = 0; p < 3; ++p) {
    const float* wg = wmat[p] + (size_t)g * D_ * Cg_;   // [D, Cg] row-major
    v8f acc[8];
    #pragma unroll
    for (int i = 0; i < 8; ++i) acc[i] = vz;

    #pragma unroll
    for (int ks = 0; ks < 2; ++ks) {                    // K = Cg = 64 -> 2 steps
      v16h bf;                                           // B = X[c, n]
      #pragma unroll
      for (int i = 0; i < 16; ++i)
        bf[i] = (_Float16)xg[(size_t)(ks * 32 + kbB + i) * N_ + nw + ncol];
      #pragma unroll
      for (int dt = 0; dt < 8; ++dt) {                   // 8 d-tiles of 16
        v16h af;                                          // A = W[d, c]
        const float* ar = wg + (size_t)(dt * 16 + mlane) * Cg_ + ks * 32 + kbB;
        #pragma unroll
        for (int i = 0; i < 8; ++i) { af[i] = (_Float16)ar[i]; af[8 + i] = (_Float16)ar[8 + i]; }
        acc[dt] = wmma16(af, bf, acc[dt]);
      }
    }

    // epilogue: bias + BN(eval) + exact GELU, store transposed [n, d] f16
    const float* bnp = bnm[p];                           // [4, G, D]
    _Float16* outb = outp[p] + ((size_t)bg * N_ + nw + ncol) * D_;
    #pragma unroll
    for (int dt = 0; dt < 8; ++dt) {
      #pragma unroll
      for (int r = 0; r < 8; ++r) {
        const int d = dt * 16 + mbase + r;
        float y = acc[dt][r] + bias[p][g * D_ + d];
        const float gam = bnp[(0 * G_ + g) * D_ + d];
        const float bet = bnp[(1 * G_ + g) * D_ + d];
        const float mea = bnp[(2 * G_ + g) * D_ + d];
        const float var = bnp[(3 * G_ + g) * D_ + d];
        y = (y - mea) * (gam * rsqrtf(var + EPS)) + bet;
        outb[d] = (_Float16)gelu_exact(y);
      }
    }
  }
}

// ---------------------------------------------------------------------------
// Kernel 2: flash attention per (b,g). grid = (B*G, N/128), block = 256.
// Each wave owns 16 query rows; streams K/V in 64-row tiles through LDS;
// online softmax; P goes through per-wave-private LDS (C-layout -> A-layout).
// Output written as attT[b, n, g*128+d] f16 (n-major, fusion-GEMM friendly).
// ---------------------------------------------------------------------------
__global__ __launch_bounds__(256)
void flash_attn_kernel(const _Float16* __restrict__ qT, const _Float16* __restrict__ kT,
                       const _Float16* __restrict__ vT, _Float16* __restrict__ attT)
{
  __shared__ _Float16 Klds[64 * 128];        // 16 KB
  __shared__ _Float16 Vlds[64 * 128];        // 16 KB
  __shared__ _Float16 Plds[8 * 16 * 64];     // 16 KB (per-wave 16x64 P tile)

  const int bg  = blockIdx.x, b = bg >> 3, g = bg & 7;
  const int q0  = blockIdx.y * 128;
  const int tid = threadIdx.x, w = tid >> 5, lane = tid & 31;
  const int mlane = lane & 15;
  const int kbA   = (lane >> 4) << 3;
  const int kbB   = (lane >> 4) << 4;
  const int mbase = (lane >> 4) << 3;
  const int ncol  = lane & 15;
  const float scale = 0.08838834764831845f;  // 1/sqrt(128)

  // Preload this wave's 16 query rows as A fragments (K = D = 128 -> 4 steps)
  v16h qf[4];
  {
    const _Float16* qr = qT + ((size_t)bg * N_ + q0 + w * 16 + mlane) * D_;
    #pragma unroll
    for (int ks = 0; ks < 4; ++ks) {
      #pragma unroll
      for (int i = 0; i < 8; ++i) {
        qf[ks][i]     = qr[ks * 32 + kbB + i];
        qf[ks][8 + i] = qr[ks * 32 + kbB + 8 + i];
      }
    }
  }

  const v8f vz = {};
  v8f oacc[8];                               // O[16 rows, 128 d] accumulator
  #pragma unroll
  for (int i = 0; i < 8; ++i) oacc[i] = vz;
  float mrun[8], lrun[8];
  #pragma unroll
  for (int r = 0; r < 8; ++r) { mrun[r] = -1e30f; lrun[r] = 0.0f; }

  _Float16* pw = Plds + w * (16 * 64);       // private P region for this wave

  for (int mt = 0; mt < 16; ++mt) {          // 16 key tiles of 64 rows
    const size_t moff = ((size_t)bg * N_ + mt * 64) * D_;
    const uint4* ksrc = (const uint4*)(kT + moff);
    const uint4* vsrc = (const uint4*)(vT + moff);
    #pragma unroll
    for (int i = 0; i < 4; ++i) {            // 64*128 halves each, vec8 copies
      ((uint4*)Klds)[tid + i * 256] = ksrc[tid + i * 256];
      ((uint4*)Vlds)[tid + i * 256] = vsrc[tid + i * 256];
    }
    if (mt + 1 < 16) __builtin_prefetch((const void*)(kT + moff + 64 * 128), 0, 1);
    __syncthreads();

    // S[16, 64] = scale * Q Kt   (B[k=d, col=m] = Klds[m*128 + d], contiguous)
    v8f s[4];
    #pragma unroll
    for (int ct = 0; ct < 4; ++ct) {
      s[ct] = vz;
      #pragma unroll
      for (int ks = 0; ks < 4; ++ks) {
        v16h bf;
        const _Float16* br = Klds + (size_t)(ct * 16 + ncol) * 128 + ks * 32 + kbB;
        #pragma unroll
        for (int i = 0; i < 16; ++i) bf[i] = br[i];
        s[ct] = wmma16(qf[ks], bf, s[ct]);
      }
      #pragma unroll
      for (int e = 0; e < 8; ++e) s[ct][e] *= scale;
    }

    // Online softmax: row (mbase+r) lives in a 16-lane half-wave.
    #pragma unroll
    for (int r = 0; r < 8; ++r) {
      float lm = s[0][r];
      #pragma unroll
      for (int ct = 1; ct < 4; ++ct) lm = fmaxf(lm, s[ct][r]);
      #pragma unroll
      for (int mask = 1; mask < 16; mask <<= 1) lm = fmaxf(lm, __shfl_xor(lm, mask, 32));
      const float mnew  = fmaxf(mrun[r], lm);
      const float alpha = __expf(mrun[r] - mnew);
      float psum = 0.0f;
      #pragma unroll
      for (int ct = 0; ct < 4; ++ct) {
        const float pv = __expf(s[ct][r] - mnew);
        s[ct][r] = pv;
        psum += pv;
      }
      #pragma unroll
      for (int mask = 1; mask < 16; mask <<= 1) psum += __shfl_xor(psum, mask, 32);
      lrun[r] = lrun[r] * alpha + psum;
      mrun[r] = mnew;
      #pragma unroll
      for (int dt = 0; dt < 8; ++dt) oacc[dt][r] *= alpha;
    }

    // P (C-layout) -> LDS f16; same-wave LDS ops are in-order, no barrier.
    #pragma unroll
    for (int ct = 0; ct < 4; ++ct)
      #pragma unroll
      for (int r = 0; r < 8; ++r)
        pw[(mbase + r) * 64 + ct * 16 + ncol] = (_Float16)s[ct][r];

    // O += P V  (A from Plds, B[k=m, col=d] = Vlds[m*128 + d])
    #pragma unroll
    for (int ks2 = 0; ks2 < 2; ++ks2) {
      v16h af;
      const _Float16* ar = pw + mlane * 64 + ks2 * 32 + kbB;
      #pragma unroll
      for (int i = 0; i < 8; ++i) { af[i] = ar[i]; af[8 + i] = ar[8 + i]; }
      #pragma unroll
      for (int dt = 0; dt < 8; ++dt) {
        v16h bf;
        const _Float16* br = Vlds + (size_t)(ks2 * 32 + kbB) * 128 + dt * 16 + ncol;
        #pragma unroll
        for (int i = 0; i < 16; ++i) bf[i] = br[(size_t)i * 128];
        oacc[dt] = wmma16(af, bf, oacc[dt]);
      }
    }
    __syncthreads();                         // before next tile overwrites K/V
  }

  // Normalize by row-sum; write attT[b, n, g*128 + d] f16.
  _Float16* ob = attT + ((size_t)b * N_ + q0 + w * 16) * E_ + g * D_;
  #pragma unroll
  for (int r = 0; r < 8; ++r) {
    const float inv = 1.0f / lrun[r];
    #pragma unroll
    for (int dt = 0; dt < 8; ++dt)
      ob[(size_t)(mbase + r) * E_ + dt * 16 + ncol] = (_Float16)(oacc[dt][r] * inv);
  }
}

// ---------------------------------------------------------------------------
// Kernel 3: fusion GEMM [C,2C]x[2C,n] + bias + BN + GELU + channel LayerNorm.
// grid = (B, N/16), block = 256; each wave: 64(c) x 16(n); LN done in-block
// over the full 512-channel column parked in LDS.
// ---------------------------------------------------------------------------
__global__ __launch_bounds__(256)
void fuse_ln_kernel(const _Float16* __restrict__ attT,
                    const float* __restrict__ w_f, const float* __restrict__ b_f,
                    const float* __restrict__ bn_f,
                    const float* __restrict__ ln_g, const float* __restrict__ ln_b,
                    float* __restrict__ out)
{
  __shared__ float ytile[C_ * 16];           // 32 KB
  __shared__ float psum[16][16], psq[16][16];
  __shared__ float mu_s[16], rs_s[16];

  const int b   = blockIdx.x;
  const int n0  = blockIdx.y * 16;
  const int tid = threadIdx.x, w = tid >> 5, lane = tid & 31;
  const int mlane = lane & 15;
  const int kbA   = (lane >> 4) << 3;
  const int kbB   = (lane >> 4) << 4;
  const int mbase = (lane >> 4) << 3;
  const int ncol  = lane & 15;
  const int c0    = w * 64;

  const v8f vz = {};
  v8f acc[4];
  #pragma unroll
  for (int i = 0; i < 4; ++i) acc[i] = vz;

  const _Float16* bbase = attT + ((size_t)b * N_ + n0 + ncol) * E_;

  for (int k0 = 0; k0 < E_; k0 += 32) {      // K = 2C = 1024 -> 32 steps
    v16h bf;                                  // B[k=e, col=n] contiguous in e
    const _Float16* br = bbase + k0 + kbB;
    #pragma unroll
    for (int i = 0; i < 16; ++i) bf[i] = br[i];
    #pragma unroll
    for (int rt = 0; rt < 4; ++rt) {
      v16h af;                                // A = w_f[c, e] (f32 -> f16)
      const float* ar = w_f + (size_t)(c0 + rt * 16 + mlane) * E_ + k0 + kbB;
      #pragma unroll
      for (int i = 0; i < 8; ++i) { af[i] = (_Float16)ar[i]; af[8 + i] = (_Float16)ar[8 + i]; }
      acc[rt] = wmma16(af, bf, acc[rt]);
    }
  }

  // bias + BN + GELU, park tile in LDS for the channel LayerNorm
  #pragma unroll
  for (int rt = 0; rt < 4; ++rt) {
    #pragma unroll
    for (int r = 0; r < 8; ++r) {
      const int c = c0 + rt * 16 + mbase + r;
      float y = acc[rt][r] + b_f[c];
      y = (y - bn_f[2 * C_ + c]) * (bn_f[c] * rsqrtf(bn_f[3 * C_ + c] + EPS)) + bn_f[C_ + c];
      ytile[c * 16 + ncol] = gelu_exact(y);
    }
  }
  __syncthreads();

  // LayerNorm over 512 channels for 16 columns: 16 segs x 16 cols partials
  {
    const int col = tid & 15, seg = tid >> 4;
    float s0 = 0.0f, s1 = 0.0f;
    for (int c = seg * 32; c < seg * 32 + 32; ++c) {
      const float v = ytile[c * 16 + col];
      s0 += v; s1 += v * v;
    }
    psum[seg][col] = s0; psq[seg][col] = s1;
  }
  __syncthreads();
  if (tid < 16) {
    float s0 = 0.0f, s1 = 0.0f;
    for (int seg = 0; seg < 16; ++seg) { s0 += psum[seg][tid]; s1 += psq[seg][tid]; }
    const float mu  = s0 * (1.0f / C_);
    const float var = s1 * (1.0f / C_) - mu * mu;
    mu_s[tid] = mu;
    rs_s[tid] = rsqrtf(var + EPS);
  }
  __syncthreads();
  for (int i = 0; i < 32; ++i) {
    const int idx = tid + i * 256;           // 0..8191
    const int c = idx >> 4, col = idx & 15;
    out[((size_t)b * C_ + c) * N_ + n0 + col] =
        (ytile[c * 16 + col] - mu_s[col]) * rs_s[col] * ln_g[c] + ln_b[c];
  }
}

// ---------------------------------------------------------------------------
extern "C" void kernel_launch(void* const* d_in, const int* in_sizes, int n_in,
                              void* d_out, int out_size, void* d_ws, size_t ws_size,
                              hipStream_t stream) {
  const float* x   = (const float*)d_in[0];
  const float* wq  = (const float*)d_in[1];
  const float* bq  = (const float*)d_in[2];
  const float* bnq = (const float*)d_in[3];
  const float* wk  = (const float*)d_in[4];
  const float* bk  = (const float*)d_in[5];
  const float* bnk = (const float*)d_in[6];
  const float* wv  = (const float*)d_in[7];
  const float* bv  = (const float*)d_in[8];
  const float* bnv = (const float*)d_in[9];
  const float* wf  = (const float*)d_in[10];
  const float* bfc = (const float*)d_in[11];
  const float* bnf = (const float*)d_in[12];
  const float* lng = (const float*)d_in[13];
  const float* lnb = (const float*)d_in[14];
  float* out = (float*)d_out;

  // Workspace: qT | kT | vT ([B,G,N,D] f16, 16 MB each) + attT ([B,N,2C] f16,
  // 16 MB) = 64 MB total.
  const size_t QKV = (size_t)B_ * G_ * N_ * D_;
  _Float16* qT   = (_Float16*)d_ws;
  _Float16* kT   = qT + QKV;
  _Float16* vT   = kT + QKV;
  _Float16* attT = vT + QKV;

  proj_qkv_kernel<<<dim3(B_ * G_, N_ / 128), 256, 0, stream>>>(
      x, wq, bq, bnq, wk, bk, bnk, wv, bv, bnv, qT, kT, vT);
  flash_attn_kernel<<<dim3(B_ * G_, N_ / 128), 256, 0, stream>>>(qT, kT, vT, attT);
  fuse_ln_kernel<<<dim3(B_, N_ / 16), 256, 0, stream>>>(
      attT, wf, bfc, bnf, lng, lnb, out);
}
